// TinyDecoderLM_58222576664902
// MI455X (gfx1250) — matrix operand order repaired and checked by reading, not compiled, not run on hardware
//
#include <hip/hip_runtime.h>

// ---------------- problem constants (TinyDecoderLM) ----------------
#define BB   2
#define SS   1024
#define DDm  512
#define HHn  8
#define LLl  4
#define VVv  32000
#define HDd  64
#define MMr  (BB * SS)          // 2048 token rows
#define ASCALE 0.125f           // 1/sqrt(64)

typedef __attribute__((ext_vector_type(8)))  float  v8f;
typedef __attribute__((ext_vector_type(8)))  int    v8i_t;
typedef __attribute__((ext_vector_type(16))) __bf16 v16bf;
typedef unsigned int u32x4 __attribute__((ext_vector_type(4)));
typedef int          i32x8 __attribute__((ext_vector_type(8)));
typedef int          i32x4 __attribute__((ext_vector_type(4)));

// Tensor Data Mover availability (this toolchain: 6-arg builtin, therock-10.0 style)
#if defined(__HIP_DEVICE_COMPILE__) && defined(__gfx1250__)
#if __has_builtin(__builtin_amdgcn_tensor_load_to_lds)
#define USE_TDM 1
#endif
#endif
#ifndef USE_TDM
#define USE_TDM 0
#endif

__device__ __forceinline__ unsigned short f2bfbits(float f) {
  unsigned u = __builtin_bit_cast(unsigned, f);
  unsigned r = u + 0x7FFFu + ((u >> 16) & 1u);   // round-to-nearest-even
  return (unsigned short)(r >> 16);
}
__device__ __forceinline__ float bf2f(unsigned short h) {
  unsigned u = ((unsigned)h) << 16;
  return __builtin_bit_cast(float, u);
}
__device__ __forceinline__ v8f wmma_bf16(v8i_t a, v8i_t b, v8f c) {
  return __builtin_amdgcn_wmma_f32_16x16x32_bf16(
      false, __builtin_bit_cast(v16bf, a),
      false, __builtin_bit_cast(v16bf, b),
      (short)0, c, false, false);
}

#if USE_TDM
// 2D tile DMA: global (row-major, dword units) -> LDS, via Tensor Data Mover.
// D# per CDNA5 ISA ch.8: group0 = {count/lds_addr/global_addr/type}, group1 =
// {mask, data_size, tensor dims, tile dims, dim0 stride}. data_size=4B.
__device__ __forceinline__ void tdm_load_2d(unsigned lds_off, const void* gptr,
                                            unsigned rowlen_dw, unsigned nrows,
                                            unsigned tensor_w_dw, unsigned tensor_h,
                                            unsigned stride_dw) {
  unsigned long long ga = (unsigned long long)(uintptr_t)gptr;
  u32x4 g0;
  g0[0] = 1u;                                        // count=1, user desc
  g0[1] = lds_off;                                   // lds_addr (bytes)
  g0[2] = (unsigned)(ga & 0xFFFFFFFFu);              // global_addr[31:0]
  g0[3] = (unsigned)((ga >> 32) & 0x01FFFFFFu) | (2u << 30);  // ga[56:32] | type=2
  i32x8 g1;
  g1[0] = (int)(2u << 16);                           // workgroup_mask=0, data_size=2 (4B)
  g1[1] = (int)((tensor_w_dw & 0xFFFFu) << 16);      // [63:48] tensor_dim0 lo16
  g1[2] = (int)((tensor_w_dw >> 16) | ((tensor_h & 0xFFFFu) << 16));
  g1[3] = (int)((tensor_h >> 16) | (rowlen_dw << 16));     // [127:112] tile_dim0
  g1[4] = (int)nrows;                                // [143:128] tile_dim1 (tile_dim2=0)
  g1[5] = (int)stride_dw;                            // tensor_dim0_stride[31:0]
  g1[6] = 0;
  g1[7] = 0;
  i32x4 z4; z4[0] = 0; z4[1] = 0; z4[2] = 0; z4[3] = 0;   // groups 2/3 unused (<=2D)
  i32x8 z8;
  for (int i = 0; i < 8; ++i) z8[i] = 0;
  __builtin_amdgcn_tensor_load_to_lds(g0, g1, z4, z4, z8, 0);
}
__device__ __forceinline__ unsigned lds_off32(const void* p) {
  return (unsigned)(uintptr_t)p;   // LDS aperture: offset lives in low 32 bits
}
#endif

// ---------------- f32 [K,N] -> bf16 [N,K] transposed weight staging ----------------
// 64x64 tiles through LDS; coalesced reads and writes. Output is the
// fragment-ready B layout so the GEMM needs no per-k-step repack.
__global__ __launch_bounds__(256) void cvt_t_bf16(const float* __restrict__ src,
                                                  unsigned short* __restrict__ dst,
                                                  int K, int N) {
  __shared__ unsigned short tile[64][65];   // +1 pad: avoid bank conflicts
  const int bk = blockIdx.y * 64, bn = blockIdx.x * 64;
  const int tid = threadIdx.x;
  for (int i = tid; i < 4096; i += 256) {
    int r = i >> 6, c = i & 63;            // r = k, c = n (coalesced over n)
    tile[c][r] = f2bfbits(src[(size_t)(bk + r) * N + bn + c]);
  }
  __syncthreads();
  for (int i = tid; i < 4096; i += 256) {
    int r = i >> 6, c = i & 63;            // r = n, c = k (coalesced over k)
    dst[(size_t)(bn + r) * K + bk + c] = tile[r][c];
  }
}

// ---------------- embedding gather: x = tok_emb[tok] + pos_emb ----------------
__global__ __launch_bounds__(256) void embed_k(const int* __restrict__ tokens,
                                               const float* __restrict__ tok_emb,
                                               const float* __restrict__ pos_emb,
                                               float* __restrict__ x) {
  int row = blockIdx.x, tid = threadIdx.x;
  int tok = tokens[row];
  int sp  = row % SS;
  float* xr = x + (size_t)row * DDm;
  const float* te = tok_emb + (size_t)tok * DDm;
  const float* pe = pos_emb + (size_t)sp * DDm;
  xr[tid]       = te[tid]       + pe[tid];
  xr[tid + 256] = te[tid + 256] + pe[tid + 256];
}

// ---------------- LayerNorm (f32 in, bf16 out), one row per 256-thread block ----------------
__global__ __launch_bounds__(256) void ln_bf16(const float* __restrict__ x,
                                               const float* __restrict__ g,
                                               const float* __restrict__ b,
                                               unsigned short* __restrict__ out) {
  __shared__ float red[256];
  int row = blockIdx.x, tid = threadIdx.x;
  const float* xr = x + (size_t)row * DDm;
  float a0 = xr[tid], a1 = xr[tid + 256];
  red[tid] = a0 + a1;
  __syncthreads();
  for (int off = 128; off > 0; off >>= 1) {
    if (tid < off) red[tid] += red[tid + off];
    __syncthreads();
  }
  float mean = red[0] * (1.0f / (float)DDm);
  __syncthreads();
  float d0 = a0 - mean, d1 = a1 - mean;
  red[tid] = d0 * d0 + d1 * d1;
  __syncthreads();
  for (int off = 128; off > 0; off >>= 1) {
    if (tid < off) red[tid] += red[tid + off];
    __syncthreads();
  }
  float rstd = rsqrtf(red[0] * (1.0f / (float)DDm) + 1e-5f);
  unsigned short* o = out + (size_t)row * DDm;
  o[tid]       = f2bfbits(d0 * rstd * g[tid]       + b[tid]);
  o[tid + 256] = f2bfbits(d1 * rstd * g[tid + 256] + b[tid + 256]);
}

// ---------------- BF16 WMMA GEMM: C[M,N] = A[M,K] @ Bt[N,K]^T (+bias, gelu, residual) ----
// Block tile 128x128, K-tile 32, 256 threads = 8 waves (4x2); wave tile 32x64.
// Both A ([M,K]) and Bt ([N,K], pre-transposed weights) tiles are identical
// 2D strided tiles -> each fed by one TDM descriptor (waves 0 and 1 issue);
// all fragment loads are contiguous b128 LDS reads. Zero per-k repack VALU.
__global__ __launch_bounds__(256) void gemm_bf16_wmma(
    const unsigned short* __restrict__ A, const unsigned short* __restrict__ Bt,
    const float* __restrict__ bias, const float* __restrict__ residual,
    float* __restrict__ outF, unsigned short* __restrict__ outB,
    int Mdim, int Ndim, int Kdim, int gelu) {
  (void)Mdim;
  __shared__ unsigned short lA[128 * 32];    // [m][k]
  __shared__ unsigned short lBT[128 * 32];   // [n][k]
  const int tid = threadIdx.x;
  const int lane = tid & 31, wid = tid >> 5;
  const int wm = (wid >> 1) * 32, wn = (wid & 1) * 64;
  const int bm = blockIdx.y * 128, bn = blockIdx.x * 128;
  const int half = lane >> 4, lm16 = lane & 15;

  v8f acc[2][4];
  for (int i = 0; i < 2; ++i)
    for (int j = 0; j < 4; ++j)
      for (int r = 0; r < 8; ++r) acc[i][j][r] = 0.0f;

  const unsigned* Ag = (const unsigned*)A;
  const unsigned* Bg = (const unsigned*)Bt;
  unsigned* lAu  = (unsigned*)lA;
  unsigned* lBTu = (unsigned*)lBT;
  const int Ku = Kdim >> 1;

  for (int k0 = 0; k0 < Kdim; k0 += 32) {
    __syncthreads();
#if USE_TDM
    if (wid == 0) {        // A tile: 128 rows x 16 dwords, row stride Ku dwords
      tdm_load_2d(lds_off32(lA), Ag + (size_t)bm * Ku + (k0 >> 1),
                  16u, 128u, (unsigned)Ku, (unsigned)MMr, (unsigned)Ku);
    } else if (wid == 1) { // B tile: identical shape against [N,K]
      tdm_load_2d(lds_off32(lBT), Bg + (size_t)bn * Ku + (k0 >> 1),
                  16u, 128u, (unsigned)Ku, (unsigned)Ndim, (unsigned)Ku);
    }
    if (wid < 2) __builtin_amdgcn_s_wait_tensorcnt(0);
#else
    for (int i = tid; i < 2048; i += 256) {
      int r = i >> 4, c = i & 15;
      lAu[i]  = Ag[(size_t)(bm + r) * Ku + (k0 >> 1) + c];
      lBTu[i] = Bg[(size_t)(bn + r) * Ku + (k0 >> 1) + c];
    }
#endif
    __syncthreads();

    v8i_t af[2];
#pragma unroll
    for (int ms = 0; ms < 2; ++ms) {
      int mrow = wm + ms * 16 + lm16;
#pragma unroll
      for (int i = 0; i < 8; ++i) {
        af[ms][i] = (int)lAu[mrow * 16 + half * 8 + i]; // contiguous -> b128
      }
    }
    v8i_t bfr[4];
#pragma unroll
    for (int ns = 0; ns < 4; ++ns) {
      int ncol = wn + ns * 16 + lm16;
#pragma unroll
      for (int i = 0; i < 8; ++i)                              // contiguous -> b128
        bfr[ns][i] = (int)lBTu[ncol * 16 + half * 8 + i];
    }
#pragma unroll
    for (int ms = 0; ms < 2; ++ms)
#pragma unroll
      for (int ns = 0; ns < 4; ++ns)
        acc[ms][ns] = wmma_bf16(af[ms], bfr[ns], acc[ms][ns]);
  }

  // epilogue: C layout row = r + 8*half, col = lane&15 within each 16x16 tile
  for (int ms = 0; ms < 2; ++ms)
    for (int ns = 0; ns < 4; ++ns)
      for (int r = 0; r < 8; ++r) {
        int row = bm + wm + ms * 16 + r + 8 * half;
        int col = bn + wn + ns * 16 + lm16;
        float v = acc[ms][ns][r];
        if (bias) v += bias[col];
        if (gelu) v = 0.5f * v * (1.0f + erff(v * 0.70710678118f)); // exact gelu
        size_t idx = (size_t)row * Ndim + col;
        if (residual) v += residual[idx];
        if (outF) outF[idx] = v;
        if (outB) outB[idx] = f2bfbits(v);
      }
}

// ---------------- fused flash attention w/ cross-layer memory KV ----------------
// 128 threads = 4 waves; each wave owns one 16-query tile of one (b,h).
// Per-wave LDS: Q 16x64, K 32x64 ([key][d]), V^T 64x32 ([d][key]), P 16x32. 44 KB/block.
__global__ __launch_bounds__(128) void attn_wmma(
    const unsigned short* __restrict__ qkv,   // [M, 3D] bf16 (q|k|v per row)
    const unsigned short* __restrict__ memk,  // [L][M, D] bf16
    const unsigned short* __restrict__ memv,  // [L][M, D] bf16
    int nmem, unsigned short* __restrict__ ctxout /* [M, D] bf16 */) {
  __shared__ unsigned short smem[4 * 5632];
  const int tid = threadIdx.x, lane = tid & 31, wid = tid >> 5;
  const int half = lane >> 4, lm16 = lane & 15;
  unsigned short* sQ  = smem + wid * 5632;
  unsigned short* sK  = sQ + 1024;
  unsigned short* sVT = sK + 2048;
  unsigned short* sP  = sVT + 2048;
  unsigned* sQu  = (unsigned*)sQ;
  unsigned* sKu  = (unsigned*)sK;
  unsigned* sVTu = (unsigned*)sVT;
  unsigned* sPu  = (unsigned*)sP;

  int gt  = blockIdx.x * 4 + wid;
  int b   = gt / (HHn * (SS / 16));
  int rem = gt % (HHn * (SS / 16));
  int h   = rem / (SS / 16);
  int t0  = (rem % (SS / 16)) * 16;

  const int rowdw = (3 * DDm) >> 1;   // qkv row stride in dwords

  // Q tile -> LDS (16 rows x 32 dwords)
#if USE_TDM
  tdm_load_2d(lds_off32(sQ),
              (const unsigned*)qkv + (size_t)(b * SS + t0) * rowdw + (h * HDd >> 1),
              32u, 16u, (unsigned)rowdw, (unsigned)MMr, (unsigned)rowdw);
#else
  for (int i = lane; i < 512; i += 32) {
    int m = i >> 5, c = i & 31;
    sQu[i] = *(const unsigned*)(qkv + (size_t)(b * SS + t0 + m) * (3 * DDm) + h * HDd + c * 2);
  }
#endif

  float mrun[8], lrun[8];
  v8f acc[4];
  for (int r = 0; r < 8; ++r) { mrun[r] = -1e30f; lrun[r] = 0.0f; }
  for (int nt = 0; nt < 4; ++nt)
    for (int r = 0; r < 8; ++r) acc[nt][r] = 0.0f;

  for (int key0 = 0; key0 < t0 + 16; key0 += 32) {
    // K tile (natural [key][d]) via TDM; V tile 2x2-transposed to [d][key]
#if USE_TDM
    tdm_load_2d(lds_off32(sK),
                (const unsigned*)qkv + (size_t)(b * SS + key0) * rowdw + ((DDm + h * HDd) >> 1),
                32u, 32u, (unsigned)rowdw, (unsigned)MMr, (unsigned)rowdw);
#else
    for (int i = lane; i < 1024; i += 32) {
      int kk = i >> 5, c = i & 31;
      sKu[i] = *(const unsigned*)(qkv + (size_t)(b * SS + key0 + kk) * (3 * DDm) + DDm + h * HDd + c * 2);
    }
#endif
    for (int i = lane; i < 512; i += 32) {
      int k2 = i >> 5;   // 0..15 -> key = 2*k2
      int d2 = i & 31;   // 0..31 -> d = 2*d2
      const unsigned* vrow =
          (const unsigned*)qkv + (size_t)(b * SS + key0 + 2 * k2) * rowdw + ((2 * DDm + h * HDd) >> 1) + d2;
      unsigned a = vrow[0];
      unsigned c = vrow[rowdw];
      if (key0 + 32 < t0 + 16) __builtin_prefetch(vrow + 32 * (size_t)rowdw, 0, 1);
      unsigned lo = (a & 0xFFFFu) | (c << 16);
      unsigned hi = (a >> 16) | (c & 0xFFFF0000u);
      sVTu[(2 * d2) * 16 + k2]     = lo;
      sVTu[(2 * d2 + 1) * 16 + k2] = hi;
    }
#if USE_TDM
    __builtin_amdgcn_s_wait_tensorcnt(0);
#endif
    asm volatile("s_wait_loadcnt 0\n\ts_wait_dscnt 0" ::: "memory");

    // scores: S = Q @ K^T  (two 16x16 key tiles, K-dim = hd = 64 -> 2 WMMAs each)
    v8f sc0, sc1;
    for (int r = 0; r < 8; ++r) { sc0[r] = 0.0f; sc1[r] = 0.0f; }
#pragma unroll
    for (int ko = 0; ko < 64; ko += 32) {
      v8i_t aq;
#pragma unroll
      for (int i = 0; i < 8; ++i) {
        aq[i] = (int)sQu[lm16 * 32 + (ko >> 1) + half * 8 + i];
      }
      v8i_t bk0, bk1;
#pragma unroll
      for (int i = 0; i < 8; ++i) {
        int kq = (ko >> 1) + half * 8 + i;          // contiguous pairs in [key][d]
        bk0[i] = (int)sKu[lm16 * 32 + kq];
        bk1[i] = (int)sKu[(16 + lm16) * 32 + kq];
      }
      sc0 = wmma_bf16(aq, bk0, sc0);
      sc1 = wmma_bf16(aq, bk1, sc1);
    }

    // online softmax, entirely in WMMA C layout (row = r + 8*half)
    float p0[8], p1[8];
#pragma unroll
    for (int r = 0; r < 8; ++r) {
      int trow = t0 + r + 8 * half;
      float s0 = sc0[r] * ASCALE; if (key0 + lm16 > trow)      s0 = -1e30f;
      float s1 = sc1[r] * ASCALE; if (key0 + 16 + lm16 > trow) s1 = -1e30f;
      float mx = fmaxf(s0, s1);
      mx = fmaxf(mx, __shfl_xor(mx, 1, 32));
      mx = fmaxf(mx, __shfl_xor(mx, 2, 32));
      mx = fmaxf(mx, __shfl_xor(mx, 4, 32));
      mx = fmaxf(mx, __shfl_xor(mx, 8, 32));
      float mnew = fmaxf(mrun[r], mx);
      float corr = __expf(mrun[r] - mnew);
      float e0 = __expf(s0 - mnew), e1 = __expf(s1 - mnew);
      float ps = e0 + e1;
      ps += __shfl_xor(ps, 1, 32);
      ps += __shfl_xor(ps, 2, 32);
      ps += __shfl_xor(ps, 4, 32);
      ps += __shfl_xor(ps, 8, 32);
      lrun[r] = lrun[r] * corr + ps;
      mrun[r] = mnew;
#pragma unroll
      for (int nt = 0; nt < 4; ++nt) acc[nt][r] *= corr;
      p0[r] = e0; p1[r] = e1;
    }

    // bounce P through LDS to re-layout C -> A fragments (bf16)
#pragma unroll
    for (int r = 0; r < 8; ++r) {
      int m = r + 8 * half;
      sP[m * 32 + lm16]      = f2bfbits(p0[r]);
      sP[m * 32 + 16 + lm16] = f2bfbits(p1[r]);
    }
    asm volatile("s_wait_dscnt 0" ::: "memory");

    // ctx += P @ V
    v8i_t ap;
#pragma unroll
    for (int i = 0; i < 8; ++i) {
      ap[i] = (int)sPu[lm16 * 16 + half * 8 + i];
    }
#pragma unroll
    for (int nt = 0; nt < 4; ++nt) {
      v8i_t bv;
#pragma unroll
      for (int i = 0; i < 8; ++i)                    // contiguous in [d][key] -> b128
        bv[i] = (int)sVTu[(nt * 16 + lm16) * 16 + half * 8 + i];
      acc[nt] = wmma_bf16(ap, bv, acc[nt]);
    }
  }

  // cross-layer memory attention: per-token scalar keys ms[t,l] = q[t].pk[l][t]
  for (int l = 0; l < nmem; ++l) {
    const unsigned short* pk = memk + (size_t)l * MMr * DDm;
    const unsigned short* pv = memv + (size_t)l * MMr * DDm;
#pragma unroll
    for (int r = 0; r < 8; ++r) {
      int m = r + 8 * half;
      size_t tok = (size_t)(b * SS + t0 + m);
      float part = 0.0f;
#pragma unroll
      for (int j = 0; j < 4; ++j) {
        int d = lm16 * 4 + j;
        part += bf2f(sQ[m * 64 + d]) * bf2f(pk[tok * DDm + h * HDd + d]);
      }
      part += __shfl_xor(part, 1, 32);
      part += __shfl_xor(part, 2, 32);
      part += __shfl_xor(part, 4, 32);
      part += __shfl_xor(part, 8, 32);
      float s = part * ASCALE;
      float mnew = fmaxf(mrun[r], s);
      float corr = __expf(mrun[r] - mnew);
      float e = __expf(s - mnew);
      lrun[r] = lrun[r] * corr + e;
      mrun[r] = mnew;
#pragma unroll
      for (int nt = 0; nt < 4; ++nt) {
        float vv = bf2f(pv[tok * DDm + h * HDd + nt * 16 + lm16]);
        acc[nt][r] = acc[nt][r] * corr + e * vv;
      }
    }
  }

  // write ctx tile (bf16, [M, D] with head-concatenated layout)
#pragma unroll
  for (int r = 0; r < 8; ++r) {
    int m = r + 8 * half;
    size_t tok = (size_t)(b * SS + t0 + m);
    float inv = 1.0f / lrun[r];
#pragma unroll
    for (int nt = 0; nt < 4; ++nt)
      ctxout[tok * DDm + h * HDd + nt * 16 + lm16] = f2bfbits(acc[nt][r] * inv);
  }
}

// ---------------- host orchestration ----------------
extern "C" void kernel_launch(void* const* d_in, const int* in_sizes, int n_in,
                              void* d_out, int out_size, void* d_ws, size_t ws_size,
                              hipStream_t stream) {
  (void)in_sizes; (void)n_in; (void)out_size; (void)ws_size;
  const int*   tokens  = (const int*)d_in[0];
  const float* tok_emb = (const float*)d_in[1];
  const float* pos_emb = (const float*)d_in[2];
  const float* ln1_g   = (const float*)d_in[3];
  const float* ln1_b   = (const float*)d_in[4];
  const float* w_qkv   = (const float*)d_in[5];
  const float* b_qkv   = (const float*)d_in[6];
  const float* w_out   = (const float*)d_in[7];
  const float* b_out   = (const float*)d_in[8];
  const float* w_k     = (const float*)d_in[9];
  const float* b_k     = (const float*)d_in[10];
  const float* w_v     = (const float*)d_in[11];
  const float* b_v     = (const float*)d_in[12];
  const float* ln2_g   = (const float*)d_in[13];
  const float* ln2_b   = (const float*)d_in[14];
  const float* w_fc1   = (const float*)d_in[15];
  const float* b_fc1   = (const float*)d_in[16];
  const float* w_fc2   = (const float*)d_in[17];
  const float* b_fc2   = (const float*)d_in[18];
  const float* lnf_g   = (const float*)d_in[19];
  const float* lnf_b   = (const float*)d_in[20];
  const float* w_lm    = (const float*)d_in[21];

  char* ws = (char*)d_ws;
  size_t off = 0;
  auto alloc = [&](size_t bytes) -> void* {
    void* p = ws + off;
    off += bytes;
    off = (off + 255) & ~(size_t)255;
    return p;
  };
  // all weight buffers hold the TRANSPOSED [N,K] bf16 image
  unsigned short* wbqkv = (unsigned short*)alloc((size_t)LLl * DDm * 3 * DDm * 2);
  unsigned short* wbout = (unsigned short*)alloc((size_t)LLl * DDm * DDm * 2);
  unsigned short* wbk   = (unsigned short*)alloc((size_t)DDm * DDm * 2);
  unsigned short* wbv   = (unsigned short*)alloc((size_t)DDm * DDm * 2);
  unsigned short* wbfc1 = (unsigned short*)alloc((size_t)LLl * DDm * 4 * DDm * 2);
  unsigned short* wbfc2 = (unsigned short*)alloc((size_t)LLl * 4 * DDm * DDm * 2);
  unsigned short* wblm  = (unsigned short*)alloc((size_t)DDm * VVv * 2);
  float*          x     = (float*)alloc((size_t)MMr * DDm * 4);
  unsigned short* xn    = (unsigned short*)alloc((size_t)MMr * DDm * 2);
  unsigned short* qkvb  = (unsigned short*)alloc((size_t)MMr * 3 * DDm * 2);
  unsigned short* ctxb  = (unsigned short*)alloc((size_t)MMr * DDm * 2);
  unsigned short* h1b   = (unsigned short*)alloc((size_t)MMr * 4 * DDm * 2);
  unsigned short* memk  = (unsigned short*)alloc((size_t)LLl * MMr * DDm * 2);
  unsigned short* memv  = (unsigned short*)alloc((size_t)LLl * MMr * DDm * 2);

  auto cvt_t = [&](const float* s, unsigned short* dst, int K, int N) {
    dim3 grid(N / 64, K / 64);
    cvt_t_bf16<<<grid, 256, 0, stream>>>(s, dst, K, N);
  };
  for (int i = 0; i < LLl; ++i) {
    cvt_t(w_qkv + (size_t)i * DDm * 3 * DDm, wbqkv + (size_t)i * DDm * 3 * DDm, DDm, 3 * DDm);
    cvt_t(w_out + (size_t)i * DDm * DDm,     wbout + (size_t)i * DDm * DDm,     DDm, DDm);
    cvt_t(w_fc1 + (size_t)i * DDm * 4 * DDm, wbfc1 + (size_t)i * DDm * 4 * DDm, DDm, 4 * DDm);
    cvt_t(w_fc2 + (size_t)i * 4 * DDm * DDm, wbfc2 + (size_t)i * 4 * DDm * DDm, 4 * DDm, DDm);
  }
  cvt_t(w_k,  wbk,  DDm, DDm);
  cvt_t(w_v,  wbv,  DDm, DDm);
  cvt_t(w_lm, wblm, DDm, VVv);

  embed_k<<<MMr, 256, 0, stream>>>(tokens, tok_emb, pos_emb, x);

  auto gemm = [&](const unsigned short* Aa, const unsigned short* Bb, const float* bias,
                  const float* resid, float* oF, unsigned short* oB, int N, int K, int gel) {
    dim3 grid(N / 128, MMr / 128);
    gemm_bf16_wmma<<<grid, 256, 0, stream>>>(Aa, Bb, bias, resid, oF, oB, MMr, N, K, gel);
  };

  for (int i = 0; i < LLl; ++i) {
    ln_bf16<<<MMr, 256, 0, stream>>>(x, ln1_g + i * DDm, ln1_b + i * DDm, xn);
    gemm(xn, wbqkv + (size_t)i * DDm * 3 * DDm, b_qkv + i * 3 * DDm,
         nullptr, nullptr, qkvb, 3 * DDm, DDm, 0);
    gemm(xn, wbk, b_k, nullptr, nullptr, memk + (size_t)i * MMr * DDm, DDm, DDm, 0);
    gemm(xn, wbv, b_v, nullptr, nullptr, memv + (size_t)i * MMr * DDm, DDm, DDm, 0);
    attn_wmma<<<BB * HHn * (SS / 16) / 4, 128, 0, stream>>>(qkvb, memk, memv, i, ctxb);
    gemm(ctxb, wbout + (size_t)i * DDm * DDm, b_out + i * DDm, x, x, nullptr, DDm, DDm, 0);
    ln_bf16<<<MMr, 256, 0, stream>>>(x, ln2_g + i * DDm, ln2_b + i * DDm, xn);
    gemm(xn, wbfc1 + (size_t)i * DDm * 4 * DDm, b_fc1 + i * 4 * DDm,
         nullptr, nullptr, h1b, 4 * DDm, DDm, 1);
    gemm(h1b, wbfc2 + (size_t)i * 4 * DDm * DDm, b_fc2 + i * DDm, x, x, nullptr, DDm, 4 * DDm, 0);
  }
  ln_bf16<<<MMr, 256, 0, stream>>>(x, lnf_g, lnf_b, xn);
  gemm(xn, wblm, nullptr, nullptr, (float*)d_out, nullptr, VVv, DDm, 0);
}
